// SLDS_CANN_VAE_72018011619800
// MI455X (gfx1250) — matrix operand order repaired and checked by reading, not compiled, not run on hardware
//
#include <hip/hip_runtime.h>

// Problem dims
#define B_    64
#define T_    1000
#define OBS_  128
#define LAT_  16
#define KK_   8
#define CANN_ 512
#define H_    256
#define DT_TAU 0.1f

typedef __attribute__((ext_vector_type(16))) _Float16 v16h;
typedef __attribute__((ext_vector_type(8)))  _Float16 v8h;
typedef __attribute__((ext_vector_type(8)))  float    v8f;

// global-address-space element / vector types (force global_load_b128, LOADcnt only)
typedef __attribute__((address_space(1))) const _Float16 cgf16;
typedef __attribute__((address_space(1))) const v8h      cgv8h;

__device__ __forceinline__ v16h cat8(v8h lo, v8h hi) {
  return __builtin_shufflevector(lo, hi, 0,1,2,3,4,5,6,7,8,9,10,11,12,13,14,15);
}
// A fragment (16x32 f16, MxK) from LDS. Lane l: row m=l&15; VGPRs 0-7 hold
// K=half*16+{0..15}  (half = l>>4).
__device__ __forceinline__ v16h load_a_frag(const _Float16* base, int stride,
                                            int kbase, int m16, int half) {
  const _Float16* p = base + m16 * stride + kbase + half * 16;
  v8h lo = *(const v8h*)p;
  v8h hi = *(const v8h*)(p + 8);
  return cat8(lo, hi);
}
// B fragment (32x16 f16, KxN) from global weights stored [N][K] row-major:
// lane l reads 16 contiguous f16 of row n = nt*16+(l&15) at col kt*32 + half*16.
__device__ __forceinline__ v16h load_b_frag(const cgf16* p) {
  v8h lo = *(const cgv8h*)p;
  v8h hi = *(const cgv8h*)(p + 8);
  return cat8(lo, hi);
}
__device__ __forceinline__ v8f splat8(float v) { v8f r = {v,v,v,v,v,v,v,v}; return r; }
__device__ __forceinline__ float sigf(float x) { return 1.0f / (1.0f + __expf(-x)); }
// Defeat LICM: redefine a pointer every loop iteration so loads through it are
// not loop-invariant (prevents hoist-then-spill of streamed weight fragments).
__device__ __forceinline__ const cgf16* launder(const cgf16* p) {
  asm volatile("" : "+s"(p));
  return p;
}

#define WMMA_F16(a,b,c) \
  __builtin_amdgcn_wmma_f32_16x16x32_f16(false,(a),false,(b),(short)0,(c),false,false)

// ---------------------------------------------------------------------------
// Kernel 1: convert weights to f16 WMMA-friendly buffers, combine biases.
// ---------------------------------------------------------------------------
__launch_bounds__(256)
__global__ void prep_weights(const float* __restrict__ Wih, const float* __restrict__ Whh,
                             const float* __restrict__ bih, const float* __restrict__ bhh,
                             const float* __restrict__ Wi,  const float* __restrict__ J,
                             const float* __restrict__ We,
                             _Float16* __restrict__ Wih16, _Float16* __restrict__ Whh16,
                             float* __restrict__ biasc,
                             _Float16* __restrict__ Wi16p, _Float16* __restrict__ J16,
                             _Float16* __restrict__ We16) {
  int i = blockIdx.x * 256 + threadIdx.x;
  if (i < 1024 * OBS_) Wih16[i] = (_Float16)Wih[i];
  if (i < 1024 * H_)   Whh16[i] = (_Float16)Whh[i];
  if (i < 1024)        biasc[i] = bih[i] + bhh[i];
  if (i < CANN_ * CANN_) J16[i] = (_Float16)J[i];
  if (i < OBS_ * CANN_)  We16[i] = (_Float16)We[i];
  if (i < CANN_ * 32) {               // Wi padded [512][32], K 16..31 zero
    int r = i >> 5, c = i & 31;
    Wi16p[i] = (c < LAT_) ? (_Float16)Wi[r * LAT_ + c] : (_Float16)0.f;
  }
}

// ---------------------------------------------------------------------------
// Kernel 2: LSTM scan. 4 WGs x 256 threads; WG owns 16 batch rows, loops T.
// Wave w owns hidden cols [32w,32w+32) across all four gates -> wave-local
// gate combine; cell state c lives in VGPRs; h double-buffered in LDS f16.
// Weights streamed from L2 every step (laundered global pointers).
// ---------------------------------------------------------------------------
__launch_bounds__(256)
__global__ void lstm_scan(const float* __restrict__ y,
                          const _Float16* __restrict__ Wih16,
                          const _Float16* __restrict__ Whh16,
                          const float* __restrict__ biasc,
                          _Float16* __restrict__ rnn16) {
  __shared__ __align__(16) _Float16 xs[2][16][OBS_];  // 8 KB
  __shared__ __align__(16) _Float16 hs[2][16][H_];    // 16 KB
  const int wg   = blockIdx.x;
  const int tid  = threadIdx.x;
  const int lane = tid & 31;
  const int wv   = tid >> 5;          // 0..7
  const int m16  = lane & 15;
  const int half = lane >> 4;

  // n-tiles: [i0,i1,f0,f1,g0,g1,o0,o1]
  int ntl[8];
#pragma unroll
  for (int g = 0; g < 4; ++g) { ntl[2*g] = g*16 + 2*wv; ntl[2*g+1] = g*16 + 2*wv + 1; }
  float biasv[8];
#pragma unroll
  for (int j = 0; j < 8; ++j) biasv[j] = biasc[ntl[j]*16 + m16];
  // per-lane element offsets of this wave's fragment rows within each weight matrix
  size_t offIh[8], offHh[8];
#pragma unroll
  for (int j = 0; j < 8; ++j) {
    offIh[j] = (size_t)(ntl[j]*16 + m16) * OBS_ + half*16;
    offHh[j] = (size_t)(ntl[j]*16 + m16) * H_   + half*16;
  }
  const cgf16* WihG = (const cgf16*)Wih16;
  const cgf16* WhhG = (const cgf16*)Whh16;

  for (int i = tid; i < 16 * H_; i += 256) ((_Float16*)hs[0])[i] = (_Float16)0.f;
  v8f cregs[2] = { splat8(0.f), splat8(0.f) };

  for (int t = 0; t < T_; ++t) {
    const int pb = t & 1, nb = pb ^ 1;
    const cgf16* WihL = launder(WihG);   // fresh every iter: no LICM hoist
    const cgf16* WhhL = launder(WhhG);
    // stage x_t (f32 -> f16) into LDS
    for (int i = tid; i < 16 * OBS_; i += 256) {
      int mm = i >> 7, cc = i & 127;
      xs[pb][mm][cc] = (_Float16)y[(((size_t)(wg*16 + mm)) * T_ + t) * OBS_ + cc];
    }
    __syncthreads();

    v8f acc[8];
#pragma unroll
    for (int j = 0; j < 8; ++j) acc[j] = splat8(biasv[j]);

    for (int kt = 0; kt < 4; ++kt) {            // x @ W_ih^T   (K=128)
      v16h a = load_a_frag(&xs[pb][0][0], OBS_, kt*32, m16, half);
      v16h b[8];
#pragma unroll
      for (int j = 0; j < 8; ++j) b[j] = load_b_frag(WihL + offIh[j] + kt*32);
#pragma unroll
      for (int j = 0; j < 8; ++j) acc[j] = WMMA_F16(a, b[j], acc[j]);
    }
    for (int kt = 0; kt < 8; ++kt) {            // h @ W_hh^T   (K=256)
      v16h a = load_a_frag(&hs[pb][0][0], H_, kt*32, m16, half);
      v16h b[8];
#pragma unroll
      for (int j = 0; j < 8; ++j) b[j] = load_b_frag(WhhL + offHh[j] + kt*32);
#pragma unroll
      for (int j = 0; j < 8; ++j) acc[j] = WMMA_F16(a, b[j], acc[j]);
    }

    // wave-local gate combine: c = sig(f)*c + sig(i)*tanh(g); h = sig(o)*tanh(c)
#pragma unroll
    for (int a = 0; a < 2; ++a) {
#pragma unroll
      for (int r = 0; r < 8; ++r) {
        float ig = sigf(acc[0 + a][r]);
        float fg = sigf(acc[2 + a][r]);
        float gg = tanhf(acc[4 + a][r]);
        float og = sigf(acc[6 + a][r]);
        float cN = fg * cregs[a][r] + ig * gg;
        cregs[a][r] = cN;
        float hN = og * tanhf(cN);
        int mm = r + 8 * half;
        int jc = 32 * wv + a * 16 + m16;
        _Float16 h16 = (_Float16)hN;
        hs[nb][mm][jc] = h16;
        rnn16[(((size_t)(wg*16 + mm)) * T_ + t) * H_ + jc] = h16;
      }
    }
    // one barrier per step: next iteration's top of loop only writes xs[nb]
  }
}

// ---------------------------------------------------------------------------
// Kernel 3: posterior heads + rsample + Gumbel argmax. One wave per (b,t) row.
// ---------------------------------------------------------------------------
__launch_bounds__(256)
__global__ void heads(const _Float16* __restrict__ rnn16,
                      const float* __restrict__ Wz,  const float* __restrict__ bz,
                      const float* __restrict__ Wlv, const float* __restrict__ blv,
                      const float* __restrict__ Wk,  const float* __restrict__ bk,
                      const float* __restrict__ eps, const float* __restrict__ gumbel,
                      float* __restrict__ z_out, _Float16* __restrict__ zs16,
                      int* __restrict__ k_out) {
  const int row  = blockIdx.x * 8 + (threadIdx.x >> 5);   // 0..63999
  const int lane = threadIdx.x & 31;
  const int l16  = lane & 15;
  const _Float16* h = rnn16 + (size_t)row * H_;
  const float* W1 = (lane < 16) ? (Wz + l16 * H_) : (Wlv + l16 * H_);
  float a1 = 0.f, ak = 0.f;
  for (int k = 0; k < H_; ++k) {
    float hv = (float)h[k];
    a1 += hv * W1[k];
    if (lane < KK_) ak += hv * Wk[lane * H_ + k];
  }
  a1 += (lane < 16) ? bz[l16] : blv[l16];
  float lv = __shfl(a1, l16 + 16, 32);                    // logvar to lanes 0..15
  if (lane < 16) {
    float zs = a1 + __expf(0.5f * lv) * eps[(size_t)row * LAT_ + l16];
    z_out[(size_t)row * LAT_ + l16] = zs;
    zs16 [(size_t)row * LAT_ + l16] = (_Float16)zs;
  }
  if (lane < KK_) {
    float kv = ak + bk[lane] + gumbel[(size_t)row * KK_ + lane];
    int   ki = lane;
#pragma unroll
    for (int off = 4; off >= 1; off >>= 1) {
      float ov = __shfl_xor(kv, off, 32);
      int   oi = __shfl_xor(ki, off, 32);
      if (ov > kv || (ov == kv && oi < ki)) { kv = ov; ki = oi; }
    }
    if (lane == 0) k_out[row] = ki;
  }
}

// ---------------------------------------------------------------------------
// Kernel 4: CANN decoder scan. 4 WGs x 256 threads; u in VGPRs (static lane
// ownership), s=tanh(u) double-buffered in LDS f16; We/Wi fragments pinned
// in VGPRs across all 1000 steps; J streamed from L2 each step (laundered).
// ---------------------------------------------------------------------------
__launch_bounds__(256)
__global__ void cann_scan(const _Float16* __restrict__ zs16,
                          const _Float16* __restrict__ Wi16p,  // [512][32]
                          const float* __restrict__ bi,
                          const _Float16* __restrict__ J16,    // [512][512]
                          const _Float16* __restrict__ We16,   // [128][512]
                          const float* __restrict__ be,
                          float* __restrict__ yout) {
  __shared__ __align__(16) _Float16 ss[2][16][CANN_];  // 32 KB
  __shared__ __align__(16) _Float16 zt[16][32];        // 1 KB (K padded to 32)
  const int wg   = blockIdx.x;
  const int tid  = threadIdx.x;
  const int lane = tid & 31;
  const int wv   = tid >> 5;
  const int m16  = lane & 15;
  const int half = lane >> 4;

  v16h wif[4];                                   // Wi B-frags, constant over t
#pragma unroll
  for (int a = 0; a < 4; ++a)
    wif[a] = load_b_frag((const cgf16*)Wi16p + (size_t)((4*wv + a)*16 + m16) * 32 + half*16);
  v16h wef[16];                                  // We B-frags, constant over t
#pragma unroll
  for (int kt = 0; kt < 16; ++kt)
    wef[kt] = load_b_frag((const cgf16*)We16 + (size_t)(wv*16 + m16) * CANN_ + kt*32 + half*16);
  float biv[4];
#pragma unroll
  for (int a = 0; a < 4; ++a) biv[a] = bi[(4*wv + a)*16 + m16];
  const float bev = be[wv*16 + m16];
  size_t offJ[4];
#pragma unroll
  for (int a = 0; a < 4; ++a)
    offJ[a] = (size_t)((4*wv + a)*16 + m16) * CANN_ + half*16;
  const cgf16* JG = (const cgf16*)J16;

  for (int i = tid; i < 16 * CANN_; i += 256) ((_Float16*)ss[0])[i] = (_Float16)0.f;
  for (int i = tid; i < 16 * 32;    i += 256) ((_Float16*)zt)[i]    = (_Float16)0.f;
  v8f u[4] = { splat8(0.f), splat8(0.f), splat8(0.f), splat8(0.f) };

  for (int t = 0; t < T_; ++t) {
    const int pb = t & 1, nb = pb ^ 1;
    const cgf16* JL = launder(JG);               // fresh every iter: no LICM hoist
    { int mm = tid >> 4, cc = tid & 15;          // stage z_t (one elem/thread)
      zt[mm][cc] = zs16[(((size_t)(wg*16 + mm)) * T_ + t) * LAT_ + cc]; }
    __syncthreads();

    v8f acc[4];
#pragma unroll
    for (int a = 0; a < 4; ++a) acc[a] = splat8(biv[a]);

    { // I_t = z @ Wi^T + bi  (K padded to 32)
      v16h az = load_a_frag(&zt[0][0], 32, 0, m16, half);
#pragma unroll
      for (int a = 0; a < 4; ++a) acc[a] = WMMA_F16(az, wif[a], acc[a]);
    }
    for (int kt = 0; kt < 16; ++kt) {            // r @ J^T   (K=512)
      v16h av = load_a_frag(&ss[pb][0][0], CANN_, kt*32, m16, half);
      v16h b[4];
#pragma unroll
      for (int a = 0; a < 4; ++a) b[a] = load_b_frag(JL + offJ[a] + kt*32);
#pragma unroll
      for (int a = 0; a < 4; ++a) acc[a] = WMMA_F16(av, b[a], acc[a]);
    }
    // u <- (1-dt)u + dt*(I + rJ);  s = tanh(u)
#pragma unroll
    for (int a = 0; a < 4; ++a) {
#pragma unroll
      for (int r = 0; r < 8; ++r) {
        float un = (1.0f - DT_TAU) * u[a][r] + DT_TAU * acc[a][r];
        u[a][r] = un;
        int mm = r + 8 * half;
        ss[nb][mm][(4*wv + a)*16 + m16] = (_Float16)tanhf(un);
      }
    }
    __syncthreads();                             // s_new complete for emission

    v8f e = splat8(bev);                         // y_t = s @ We^T + be
#pragma unroll
    for (int kt = 0; kt < 16; ++kt) {
      v16h av = load_a_frag(&ss[nb][0][0], CANN_, kt*32, m16, half);
      e = WMMA_F16(av, wef[kt], e);
    }
#pragma unroll
    for (int r = 0; r < 8; ++r) {
      int mm = r + 8 * half;
      yout[(((size_t)(wg*16 + mm)) * T_ + t) * OBS_ + wv*16 + m16] = e[r];
    }
  }
}

// ---------------------------------------------------------------------------
extern "C" void kernel_launch(void* const* d_in, const int* in_sizes, int n_in,
                              void* d_out, int out_size, void* d_ws, size_t ws_size,
                              hipStream_t stream) {
  (void)in_sizes; (void)n_in; (void)out_size; (void)ws_size;
  const float* y    = (const float*)d_in[0];
  const float* Wih  = (const float*)d_in[1];
  const float* Whh  = (const float*)d_in[2];
  const float* bih  = (const float*)d_in[3];
  const float* bhh  = (const float*)d_in[4];
  const float* Wz   = (const float*)d_in[5];
  const float* bz   = (const float*)d_in[6];
  const float* Wlv  = (const float*)d_in[7];
  const float* blv  = (const float*)d_in[8];
  const float* Wk   = (const float*)d_in[9];
  const float* bk   = (const float*)d_in[10];
  const float* Wi   = (const float*)d_in[11];
  const float* bi   = (const float*)d_in[12];
  const float* J    = (const float*)d_in[13];
  const float* We   = (const float*)d_in[14];
  const float* be   = (const float*)d_in[15];
  const float* eps  = (const float*)d_in[16];
  const float* gum  = (const float*)d_in[17];

  // workspace carve-up (all offsets 256B aligned)
  char* w = (char*)d_ws;
  _Float16* Wih16 = (_Float16*)(w + 0);                      // 1024*128*2
  _Float16* Whh16 = (_Float16*)(w + 262144);                 // 1024*256*2
  float*    biasc = (float*)   (w + 786432);                 // 1024*4
  _Float16* J16   = (_Float16*)(w + 790528);                 // 512*512*2
  _Float16* We16  = (_Float16*)(w + 1314816);                // 128*512*2
  _Float16* Wi16p = (_Float16*)(w + 1445888);                // 512*32*2
  _Float16* rnn16 = (_Float16*)(w + 1478656);                // 64*1000*256*2
  _Float16* zsamp16 = (_Float16*)(w + 34246656);             // 64*1000*16*2

  float* y_recon = (float*)d_out;                            // [64,1000,128]
  float* z_out   = (float*)d_out + (size_t)B_ * T_ * OBS_;   // [64,1000,16]
  int*   k_out   = (int*)((float*)d_out + (size_t)B_ * T_ * (OBS_ + LAT_)); // [64,1000]

  prep_weights<<<1024, 256, 0, stream>>>(Wih, Whh, bih, bhh, Wi, J, We,
                                         Wih16, Whh16, biasc, Wi16p, J16, We16);
  lstm_scan<<<4, 256, 0, stream>>>(y, Wih16, Whh16, biasc, rnn16);
  heads<<<(B_ * T_) / 8, 256, 0, stream>>>(rnn16, Wz, bz, Wlv, blv, Wk, bk,
                                           eps, gum, z_out, zsamp16, k_out);
  cann_scan<<<4, 256, 0, stream>>>(zsamp16, Wi16p, bi, J16, We16, be, y_recon);
}
